// CharDecoder_88347477279636
// MI455X (gfx1250) — matrix operand order repaired and checked by reading, not compiled, not run on hardware
//
#include <hip/hip_runtime.h>

// ---------------- types ----------------
typedef __attribute__((ext_vector_type(16))) __bf16        v16bf;
typedef __attribute__((ext_vector_type(8)))  float         v8f;
typedef __attribute__((ext_vector_type(4)))  unsigned int  u32x4;  // plain 128-bit vector
                                                                   // (HIP's uint4 class can't
                                                                   // assign from AS(1) lvalues)
#define GLOBAL_AS __attribute__((address_space(1)))

#define DH 512
#define DE 128
#define TM 16          // batch rows per workgroup
#define NTHREADS 256   // 8 waves (wave32)
#define A_LD 1032      // bf16 elems per A row (1024 + 8 pad -> 2064B stride, 4-bank skew)
#define HS_LD 516      // fp32 elems per hstate row (512 + 4 pad)
#define L_LD 68

// ---------------- packed-weight blob offsets (bf16 elements) ----------------
// Each 32(K)x16(N) WMMA B tile is stored as 32 lanes x 16 bf16 = 512 elems (1024B),
// block index = ntile*KT + ktile (K-inner for streaming).
constexpr long BLK     = 512;
constexpr long OFF_Z2H = 0;                   constexpr long SZ_Z2H = 4L * 32 * BLK;
constexpr long OFF_RZ0 = OFF_Z2H + SZ_Z2H;    constexpr long SZ_RZ0 = 20L * 64 * BLK;
constexpr long OFF_IN0 = OFF_RZ0 + SZ_RZ0;    constexpr long SZ_IN0 = 4L * 32 * BLK;
constexpr long SZ_HN   = 16L * 32 * BLK;
constexpr long SZ_RZ   = 32L * 64 * BLK;
constexpr long SZ_IN   = 16L * 32 * BLK;
constexpr long OFF_HN0 = OFF_IN0 + SZ_IN0;
constexpr long OFF_RZ1 = OFF_HN0 + SZ_HN;
constexpr long OFF_IN1 = OFF_RZ1 + SZ_RZ;
constexpr long OFF_HN1 = OFF_IN1 + SZ_IN;
constexpr long OFF_RZ2 = OFF_HN1 + SZ_HN;
constexpr long OFF_IN2 = OFF_RZ2 + SZ_RZ;
constexpr long OFF_HN2 = OFF_IN2 + SZ_IN;
constexpr long OFF_H2V = OFF_HN2 + SZ_HN;     constexpr long SZ_H2V = 16L * 4 * BLK;
constexpr long WPACK_ELEMS = OFF_H2V + SZ_H2V;   // ~4.2M bf16 (~8.5MB), L2-resident

// ---------------- dynamic LDS layout ----------------
constexpr int SM_H_BYTES = 3 * TM * HS_LD * 4;          // h1,h2,h3 fp32
constexpr int SM_A_OFF   = SM_H_BYTES;                  // bf16 A = [x | h]
constexpr int SM_A_BYTES = TM * A_LD * 2;
constexpr int SM_L_OFF   = SM_A_OFF + SM_A_BYTES;       // logits fp32
constexpr int SM_L_BYTES = TM * L_LD * 4;
constexpr int SM_TOK_OFF = SM_L_OFF + SM_L_BYTES;       // tok / eos / seqlen
constexpr int SM_BYTES   = SM_TOK_OFF + 3 * TM * 4;     // ~136.6 KB < 320 KB/WGP

// Force addrspace(1) so the backend emits global_load_* (LOADcnt only) instead of
// flat_load_* (LOADcnt+DScnt), decoupling the weight stream from the LDS A stream.
template <typename T>
__device__ __forceinline__ const GLOBAL_AS T* to_global(const void* p) {
  return (const GLOBAL_AS T*)(unsigned long long)p;
}

// ---------------- fragment loaders ----------------
// A (16x32 bf16, ISA layout): lane<16 -> M=lane, K chunks [k0,k0+8) & [k0+16,k0+24);
// lane>=16 -> M=lane-16, chunks shifted by +8.  (LDS -> ds_load_b128 x2)
__device__ __forceinline__ v16bf load_a_frag(const __bf16* Ab, int lane, int k0) {
  int m = lane & 15;
  int base = k0 + ((lane >> 4) << 3);
  const __bf16* p = Ab + m * A_LD + base;
  union { v16bf v; u32x4 q[2]; } u;
  u.q[0] = *(const u32x4*)(p);
  u.q[1] = *(const u32x4*)(p + 16);
  return u.v;
}
// B (32x16 bf16, ISA layout): 32B contiguous per lane (coalesced 1KB/wave).
// `base` is per-column-tile uniform; `elemoff` is a compile-time-constant block
// offset after partial unroll -> folds into the 24-bit instruction offset
// (global_load_b128 v, v[lane_addr], off offset:<const>, zero per-load VALU).
__device__ __forceinline__ v16bf load_b_frag(const __bf16* base, long elemoff, int lane) {
  const GLOBAL_AS u32x4* p = to_global<u32x4>(base) + (lane << 1);
  const GLOBAL_AS u32x4* q = (const GLOBAL_AS u32x4*)((const GLOBAL_AS char*)p + elemoff * 2);
  union { v16bf v; u32x4 w[2]; } u;
  u.w[0] = q[0];
  u.w[1] = q[1];
  return u.v;
}
__device__ __forceinline__ v8f wmma_bf16(v8f acc, v16bf a, v16bf b) {
  // D = A*B + C, fp32 accumulate
  return __builtin_amdgcn_wmma_f32_16x16x32_bf16(false, a, false, b, (short)0, acc,
                                                 false, false);
}
__device__ __forceinline__ float sigmoidf_(float x) { return 1.0f / (1.0f + __expf(-x)); }

// ---------------- fused GRU layer (one timestep, 16 rows) ----------------
// A (LDS) = [x (DIN cols) | h (512 cols)] bf16. Packed blobs:
//   rzb: K=DIN+512, N=1024 ([Wi;Wh] rows 0..1023)   -> gi+gh for r,z in one pass
//   inb: K=DIN,     N=512  (Wi rows 1024..1535)     -> i_n
//   hnb: K=512,     N=512  (Wh rows 1024..1535)     -> h_n
template <int DIN>
__device__ void gru_layer(float* hst, const __bf16* Ab,
                          const __bf16* rzb, const __bf16* inb, const __bf16* hnb,
                          const float* b_ih, const float* b_hh, int wave, int lane) {
  constexpr int KTRZ = (DIN + DH) / 32;
  constexpr int KTD  = DIN / 32;
  constexpr int KTH  = DH / 32;
  const int cl = lane & 15;
  const int mb = (lane >> 4) << 3;
  for (int ct = wave; ct < 32; ct += 8) {   // 4 column-tiles per wave
    // per-tile uniform bases: one pointer add per matrix per tile
    const __bf16* rzr = rzb + (long)ct * KTRZ * BLK;
    const __bf16* rzz = rzb + (long)(32 + ct) * KTRZ * BLK;
    const __bf16* inp = inb + (long)ct * KTD * BLK;
    const __bf16* hnp = hnb + (long)ct * KTH * BLK;
    if (ct + 8 < 32)                         // pull next tile's weights toward the WGP
      __builtin_prefetch(rzb + (long)(ct + 8) * KTRZ * BLK, 0, 3);
    v8f racc = {}; v8f zacc = {}; v8f iacc = {}; v8f hacc = {};
#pragma unroll 4
    for (int kt = 0; kt < KTRZ; ++kt) {
      v16bf a = load_a_frag(Ab, lane, kt * 32);
      racc = wmma_bf16(racc, a, load_b_frag(rzr, (long)kt * BLK, lane));
      zacc = wmma_bf16(zacc, a, load_b_frag(rzz, (long)kt * BLK, lane));
    }
#pragma unroll 4
    for (int kt = 0; kt < KTD; ++kt) {
      v16bf a = load_a_frag(Ab, lane, kt * 32);
      iacc = wmma_bf16(iacc, a, load_b_frag(inp, (long)kt * BLK, lane));
    }
#pragma unroll 4
    for (int kt = 0; kt < KTH; ++kt) {
      v16bf a = load_a_frag(Ab, lane, DIN + kt * 32);
      hacc = wmma_bf16(hacc, a, load_b_frag(hnp, (long)kt * BLK, lane));
    }
    const int col = ct * 16 + cl;
    const float rb = b_ih[col] + b_hh[col];
    const float zb = b_ih[DH + col] + b_hh[DH + col];
    const float ib = b_ih[2 * DH + col];
    const float hb = b_hh[2 * DH + col];
#pragma unroll
    for (int i = 0; i < 8; ++i) {           // same C/D layout for all 4 tiles
      const int m = mb + i;
      const float hold = hst[m * HS_LD + col];
      const float r  = sigmoidf_(racc[i] + rb);
      const float z  = sigmoidf_(zacc[i] + zb);
      const float nn = tanhf(iacc[i] + ib + r * (hacc[i] + hb));
      hst[m * HS_LD + col] = (1.0f - z) * nn + z * hold;
    }
  }
}

// ---------------- weight packer: B[k][n] = k<split ? S0[n][k] : S1[n][k-split] ----------------
__global__ void CharDecoder_88347477279636_pack(__bf16* __restrict__ dst,
                                                const float* __restrict__ S0,
                                                const float* __restrict__ S1,
                                                int split, int ld0, int ld1, int n0,
                                                int KT, long total) {
  const GLOBAL_AS float* g0 = to_global<float>(S0);
  const GLOBAL_AS float* g1 = to_global<float>(S1);
  for (long idx = (long)blockIdx.x * blockDim.x + threadIdx.x; idx < total;
       idx += (long)gridDim.x * blockDim.x) {
    const int  j    = (int)(idx & 15);
    const int  lane = (int)((idx >> 4) & 31);
    const long blk  = idx >> 9;
    const int  kt = (int)(blk % KT), nt = (int)(blk / KT);
    const int  nn = n0 + nt * 16 + (lane & 15);
    const int  k  = kt * 32 + ((lane >> 4) << 3) + (j & 7) + ((j >> 3) << 4);  // lane>=16 holds K=8..15,24..31
    const float v = (k < split) ? g0[(long)nn * ld0 + k]
                                : g1[(long)nn * ld1 + (k - split)];
    dst[idx] = (__bf16)v;
  }
}

// ---------------- persistent per-tile decoder ----------------
__global__ __launch_bounds__(NTHREADS, 1)
void CharDecoder_88347477279636_decode(
    const float* __restrict__ Z, const float* __restrict__ emb,
    const float* __restrict__ z2h_b,
    const float* __restrict__ b_ih0, const float* __restrict__ b_hh0,
    const float* __restrict__ b_ih1, const float* __restrict__ b_hh1,
    const float* __restrict__ b_ih2, const float* __restrict__ b_hh2,
    const float* __restrict__ h2v_b,
    const __bf16* __restrict__ wp,
    int* __restrict__ out, int n, const int* __restrict__ max_len_p) {
  extern __shared__ char smem[];
  float*  hst = (float*)(smem);               // [3][TM][HS_LD]
  __bf16* Ab  = (__bf16*)(smem + SM_A_OFF);   // [TM][A_LD]
  float*  lgt = (float*)(smem + SM_L_OFF);    // [TM][L_LD]
  int* tok  = (int*)(smem + SM_TOK_OFF);
  int* eosm = tok + TM;
  int* slen = eosm + TM;

  const int tid = threadIdx.x;
  const int wave = tid >> 5, lane = tid & 31;
  const int row0 = blockIdx.x * TM;
  const int ml = *max_len_p;
  int* Xg = out;                      // X_gen [n][ml]
  int* SL = out + (long)n * ml;       // seq_lens [n]

  const GLOBAL_AS float* gZ   = to_global<float>(Z);
  const GLOBAL_AS float* gemb = to_global<float>(emb);

  // ---- prologue: h0 = Z @ z2h_w.T + b  (replicated into h1,h2,h3) ----
  for (int idx = tid; idx < TM * DE; idx += NTHREADS) {
    int m = idx / DE, c = idx % DE;
    Ab[m * A_LD + c] = (__bf16)gZ[(long)(row0 + m) * DE + c];
  }
  __syncthreads();
  {
    const int cl = lane & 15, mb = (lane >> 4) << 3;
    for (int ct = wave; ct < 32; ct += 8) {
      const __bf16* z2h = wp + OFF_Z2H + (long)ct * 4 * BLK;
      v8f acc = {};
#pragma unroll
      for (int kt = 0; kt < 4; ++kt) {
        v16bf a = load_a_frag(Ab, lane, kt * 32);
        acc = wmma_bf16(acc, a, load_b_frag(z2h, (long)kt * BLK, lane));
      }
      const int col = ct * 16 + cl;
      const float bias = z2h_b[col];
#pragma unroll
      for (int i = 0; i < 8; ++i) {
        const float v = acc[i] + bias;
        const int m = mb + i;
        hst[(0 * TM + m) * HS_LD + col] = v;
        hst[(1 * TM + m) * HS_LD + col] = v;
        hst[(2 * TM + m) * HS_LD + col] = v;
      }
    }
  }
  if (tid < TM) {
    tok[tid] = 1;                 // SOS
    eosm[tid] = 0;
    slen[tid] = ml;
    Xg[(long)(row0 + tid) * ml + 0] = 1;
  }
  __syncthreads();

  // ---- timestep loop (rows are independent across workgroups: no grid sync) ----
  for (int t = 1; t < ml; ++t) {
    // layer 0: A = [emb[x_t] | h1]
    for (int idx = tid; idx < TM * (DE + DH); idx += NTHREADS) {
      int m = idx / (DE + DH), c = idx % (DE + DH);
      float v = (c < DE) ? gemb[(long)tok[m] * DE + c]
                         : hst[(0 * TM + m) * HS_LD + (c - DE)];
      Ab[m * A_LD + c] = (__bf16)v;
    }
    __syncthreads();
    gru_layer<DE>(hst + 0 * TM * HS_LD, Ab, wp + OFF_RZ0, wp + OFF_IN0, wp + OFF_HN0,
                  b_ih0, b_hh0, wave, lane);
    __syncthreads();
    // layer 1: A = [h1_new | h2]
    for (int idx = tid; idx < TM * (2 * DH); idx += NTHREADS) {
      int m = idx / (2 * DH), c = idx % (2 * DH);
      float v = (c < DH) ? hst[(0 * TM + m) * HS_LD + c]
                         : hst[(1 * TM + m) * HS_LD + (c - DH)];
      Ab[m * A_LD + c] = (__bf16)v;
    }
    __syncthreads();
    gru_layer<DH>(hst + 1 * TM * HS_LD, Ab, wp + OFF_RZ1, wp + OFF_IN1, wp + OFF_HN1,
                  b_ih1, b_hh1, wave, lane);
    __syncthreads();
    // layer 2: A = [h2_new | h3]
    for (int idx = tid; idx < TM * (2 * DH); idx += NTHREADS) {
      int m = idx / (2 * DH), c = idx % (2 * DH);
      float v = (c < DH) ? hst[(1 * TM + m) * HS_LD + c]
                         : hst[(2 * TM + m) * HS_LD + (c - DH)];
      Ab[m * A_LD + c] = (__bf16)v;
    }
    __syncthreads();
    gru_layer<DH>(hst + 2 * TM * HS_LD, Ab, wp + OFF_RZ2, wp + OFF_IN2, wp + OFF_HN2,
                  b_ih2, b_hh2, wave, lane);
    __syncthreads();
    // logits = h3 @ h2v_w.T + b (argmax == argmax(softmax), skip softmax)
    for (int idx = tid; idx < TM * DH; idx += NTHREADS) {
      int m = idx / DH, c = idx % DH;
      Ab[m * A_LD + c] = (__bf16)hst[(2 * TM + m) * HS_LD + c];
    }
    __syncthreads();
    if (wave < 4) {               // wave-uniform branch: EXEC all-ones for WMMA
      const __bf16* h2v = wp + OFF_H2V + (long)wave * 16 * BLK;
      v8f acc = {};
#pragma unroll 4
      for (int kt = 0; kt < 16; ++kt) {
        v16bf a = load_a_frag(Ab, lane, kt * 32);
        acc = wmma_bf16(acc, a, load_b_frag(h2v, (long)kt * BLK, lane));
      }
      const int col = wave * 16 + (lane & 15);
      const float bias = h2v_b[col];
      const int mb = (lane >> 4) << 3;
#pragma unroll
      for (int i = 0; i < 8; ++i) lgt[(mb + i) * L_LD + col] = acc[i] + bias;
    }
    __syncthreads();
    if (tid < TM) {
      const float* lr = lgt + tid * L_LD;
      int best = 0; float bv = lr[0];
      for (int c = 1; c < 64; ++c) { float v = lr[c]; if (v > bv) { bv = v; best = c; } }
      const int eo = eosm[tid];
      Xg[(long)(row0 + tid) * ml + t] = eo ? 0 : best;   // PAD after EOS
      if (!eo && best == 2) { slen[tid] = t + 1; eosm[tid] = 1; }
      tok[tid] = best;                                   // carry feeds raw argmax
    }
    __syncthreads();
  }
  if (tid < TM) SL[row0 + tid] = slen[tid];
}

// ---------------- host entry ----------------
extern "C" void kernel_launch(void* const* d_in, const int* in_sizes, int n_in,
                              void* d_out, int out_size, void* d_ws, size_t ws_size,
                              hipStream_t stream) {
  const float* Z     = (const float*)d_in[0];
  const float* emb   = (const float*)d_in[1];
  const float* z2h_w = (const float*)d_in[2];
  const float* z2h_b = (const float*)d_in[3];
  const float* W_ih0 = (const float*)d_in[4];
  const float* W_hh0 = (const float*)d_in[5];
  const float* b_ih0 = (const float*)d_in[6];
  const float* b_hh0 = (const float*)d_in[7];
  const float* W_ih1 = (const float*)d_in[8];
  const float* W_hh1 = (const float*)d_in[9];
  const float* b_ih1 = (const float*)d_in[10];
  const float* b_hh1 = (const float*)d_in[11];
  const float* W_ih2 = (const float*)d_in[12];
  const float* W_hh2 = (const float*)d_in[13];
  const float* b_ih2 = (const float*)d_in[14];
  const float* b_hh2 = (const float*)d_in[15];
  const float* h2v_w = (const float*)d_in[16];
  const float* h2v_b = (const float*)d_in[17];
  const int*   mlp   = (const int*)d_in[18];

  const int n = in_sizes[0] / DE;          // N rows (D_Z == 128)
  __bf16* wp = (__bf16*)d_ws;              // needs ~8.5MB of d_ws

  dim3 pg(512), pb(256);
  // z2h: K=128, N=512
  CharDecoder_88347477279636_pack<<<pg, pb, 0, stream>>>(wp + OFF_Z2H, z2h_w, z2h_w, DE, DE, DE, 0, 4, SZ_Z2H);
  // layer 0 (DIN=128)
  CharDecoder_88347477279636_pack<<<pg, pb, 0, stream>>>(wp + OFF_RZ0, W_ih0, W_hh0, DE, DE, DH, 0, 20, SZ_RZ0);
  CharDecoder_88347477279636_pack<<<pg, pb, 0, stream>>>(wp + OFF_IN0, W_ih0, W_ih0, DE, DE, DE, 1024, 4, SZ_IN0);
  CharDecoder_88347477279636_pack<<<pg, pb, 0, stream>>>(wp + OFF_HN0, W_hh0, W_hh0, DH, DH, DH, 1024, 16, SZ_HN);
  // layer 1 (DIN=512)
  CharDecoder_88347477279636_pack<<<pg, pb, 0, stream>>>(wp + OFF_RZ1, W_ih1, W_hh1, DH, DH, DH, 0, 32, SZ_RZ);
  CharDecoder_88347477279636_pack<<<pg, pb, 0, stream>>>(wp + OFF_IN1, W_ih1, W_ih1, DH, DH, DH, 1024, 16, SZ_IN);
  CharDecoder_88347477279636_pack<<<pg, pb, 0, stream>>>(wp + OFF_HN1, W_hh1, W_hh1, DH, DH, DH, 1024, 16, SZ_HN);
  // layer 2 (DIN=512)
  CharDecoder_88347477279636_pack<<<pg, pb, 0, stream>>>(wp + OFF_RZ2, W_ih2, W_hh2, DH, DH, DH, 0, 32, SZ_RZ);
  CharDecoder_88347477279636_pack<<<pg, pb, 0, stream>>>(wp + OFF_IN2, W_ih2, W_ih2, DH, DH, DH, 1024, 16, SZ_IN);
  CharDecoder_88347477279636_pack<<<pg, pb, 0, stream>>>(wp + OFF_HN2, W_hh2, W_hh2, DH, DH, DH, 1024, 16, SZ_HN);
  // h2v: K=512, N=64
  CharDecoder_88347477279636_pack<<<pg, pb, 0, stream>>>(wp + OFF_H2V, h2v_w, h2v_w, DH, DH, DH, 0, 16, SZ_H2V);

  (void)hipFuncSetAttribute((const void*)CharDecoder_88347477279636_decode,
                            hipFuncAttributeMaxDynamicSharedMemorySize, SM_BYTES);
  CharDecoder_88347477279636_decode<<<dim3(n / TM), dim3(NTHREADS), SM_BYTES, stream>>>(
      Z, emb, z2h_b, b_ih0, b_hh0, b_ih1, b_hh1, b_ih2, b_hh2, h2v_b, wp,
      (int*)d_out, n, mlp);
}
